// SpectralStep_15178414424326
// MI455X (gfx1250) — compile-verified, matrix operand order checked by reading, not run on hardware
//
#include <hip/hip_runtime.h>
#include <hip/hip_bf16.h>
#include <math.h>

#define BATCH 8
#define IN_CH 64
#define RANK  32
#define WFREQ 64
#define HDIM  256
#define WDIM  512
#define PI_F  3.14159265358979f

typedef __attribute__((ext_vector_type(16))) _Float16 v16h;
typedef __attribute__((ext_vector_type(8)))  _Float16 v8h;
typedef __attribute__((ext_vector_type(8)))  float    v8f;

static __device__ __forceinline__ float2 cmulf(float2 a, float2 b) {
    return make_float2(a.x * b.x - a.y * b.y, a.x * b.y + a.y * b.x);
}

// ---------------------------------------------------------------------------
// Stockham radix-2 FFT in LDS. NT threads cooperate on one N-point transform.
// dir = -1 forward, +1 inverse (unnormalized). Returns buffer holding result.
// Leading __syncthreads covers producer writes before stage 0; trailing one
// covers the last stage's writes.
// ---------------------------------------------------------------------------
template <int N, int NT>
static __device__ __forceinline__ float2* fft_stockham(float2* src, float2* dst,
                                                       int t, float dir) {
    int s = 0;
    for (int Ns = 1; Ns < N; Ns <<= 1, ++s) {
        __syncthreads();
        const int   msk  = Ns - 1;
        const float base = dir * PI_F / (float)Ns;   // 2*pi/(2*Ns) with sign
        for (int j = t; j < N / 2; j += NT) {
            float2 v0 = src[j];
            float2 v1 = src[j + N / 2];
            const int jm = j & msk;
            float sn, cs;
            __sincosf(base * (float)jm, &sn, &cs);
            const float2 tw = make_float2(cs * v1.x - sn * v1.y,
                                          cs * v1.y + sn * v1.x);
            const int idx = ((j >> s) << (s + 1)) | jm;
            dst[idx]      = make_float2(v0.x + tw.x, v0.y + tw.y);
            dst[idx + Ns] = make_float2(v0.x - tw.x, v0.y - tw.y);
        }
        float2* tmp = src; src = dst; dst = tmp;
    }
    __syncthreads();
    return src;
}

// ---------------------------------------------------------------------------
// Kernel A: pooling + channel-mix GEMM (WMMA f16->f32) + transcendentals.
// One block per (b,h). Produces filt[(b*64+wf)*256 + h] as complex (2 floats).
// ---------------------------------------------------------------------------
__global__ __launch_bounds__(256)
void filt_kernel(const float* __restrict__ x, const float* __restrict__ dt,
                 const float* __restrict__ w, const float* __restrict__ bias,
                 float* __restrict__ filt) {
    __shared__ _Float16 poolT[64 * 64];   // [wf][c]  (A matrix, row-major in K)
    __shared__ _Float16 wlds[64 * 64];    // [o][c]   (B matrix, row-major in K)
    __shared__ float    params[64 * 64];  // [wf][o]
    __shared__ float    red[512];

    const int t    = threadIdx.x;
    const int bh   = blockIdx.x;
    const int bidx = bh >> 8;    // / HDIM
    const int h    = bh & 255;

    // --- pooling: mean over 8 consecutive W samples --------------------------
    for (int i = 0; i < 16; ++i) {
        const int idx = t + 256 * i;          // 0..4095
        const int c   = idx >> 6;
        const int wf  = idx & 63;
        const float4* p =
            (const float4*)(x + (((size_t)(bidx * 64 + c) * 256 + h) * 512 + wf * 8));
        const float4 a0 = p[0], a1 = p[1];
        const float s = (a0.x + a0.y + a0.z + a0.w + a1.x + a1.y + a1.z + a1.w) * 0.125f;
        poolT[wf * 64 + c] = (_Float16)s;
        wlds[idx]          = (_Float16)w[idx];
    }
    __syncthreads();

    // --- GEMM: D[wf,o] = sum_c poolT[wf,c] * wlds[o,c]; 16x16 tiles ---------
    const int wave   = t >> 5;
    const int lane   = t & 31;
    const int lrow   = lane & 15;
    const int half   = lane >> 4;
    for (int tile = wave * 2; tile < wave * 2 + 2; ++tile) {
        const int mt = tile >> 2;
        const int nt = tile & 3;
        v8f acc = {};
        const _Float16* arow = &poolT[(mt * 16 + lrow) * 64];
        const _Float16* brow = &wlds[(nt * 16 + lrow) * 64];
        for (int kb = 0; kb < 2; ++kb) {
            const v8h alo = *(const v8h*)(arow + kb * 32 + half * 8);
            const v8h ahi = *(const v8h*)(arow + kb * 32 + 16 + half * 8);
            const v8h blo = *(const v8h*)(brow + kb * 32 + half * 8);
            const v8h bhi = *(const v8h*)(brow + kb * 32 + 16 + half * 8);
            v16h af, bf;
            for (int e = 0; e < 8; ++e) {
                af[e] = alo[e]; af[e + 8] = ahi[e];
                bf[e] = blo[e]; bf[e + 8] = bhi[e];
            }
            acc = __builtin_amdgcn_wmma_f32_16x16x32_f16(
                false, af, false, bf, (short)0, acc, false, false);
        }
        for (int i = 0; i < 8; ++i) {
            const int M = mt * 16 + half * 8 + i;   // wf
            const int N = nt * 16 + lrow;           // o
            params[M * 64 + N] = acc[i];
        }
    }
    __syncthreads();

    // --- softplus / tanh / exp / sincos + rank reduction --------------------
    const int   wf  = t & 63;
    const int   rg  = t >> 6;      // 4 groups of 8 ranks
    const float dtb = dt[bidx];
    float sr = 0.f, si = 0.f;
    for (int r = rg * 8; r < rg * 8 + 8; ++r) {
        const float pnu   = params[wf * 64 + r]        + bias[r];
        const float pth   = params[wf * 64 + RANK + r] + bias[RANK + r];
        const float nu    = (pnu > 20.f) ? pnu : log1pf(__expf(pnu));
        const float theta = tanhf(pth) * PI_F;
        const float decay = __expf(-nu * dtb);
        float sn, cs;
        __sincosf(theta * dtb, &sn, &cs);
        sr += decay * cs;
        si += decay * sn;
    }
    red[t]       = sr;
    red[256 + t] = si;
    __syncthreads();
    if (rg == 0) {
        const float fr = red[wf] + red[wf + 64] + red[wf + 128] + red[wf + 192];
        const float fi = red[256 + wf] + red[256 + wf + 64] +
                         red[256 + wf + 128] + red[256 + wf + 192];
        float* o = filt + ((size_t)(bidx * 64 + wf) * 256 + h) * 2;
        o[0] = fr;
        o[1] = fi;
    }
}

// ---------------------------------------------------------------------------
// Kernel B: forward 512-pt row FFT (two real rows packed per complex FFT),
// store only wf<64 bins, transposed to (plane, wf, h) for kernel C.
// ---------------------------------------------------------------------------
__global__ __launch_bounds__(64)
void rowfft_kernel(const float* __restrict__ x, float* __restrict__ sm) {
    __shared__ float2 bufA[512], bufB[512];
    const int t     = threadIdx.x;
    const int p     = blockIdx.x;          // plane*128 + rowpair
    const int plane = p >> 7;
    const int h0    = (p & 127) * 2;
    const float* x0 = x + ((size_t)plane * 256 + h0) * 512;
    const float* x1 = x0 + 512;
    for (int i = 0; i < 8; ++i) {
        const int n = t + 64 * i;
        bufA[n] = make_float2(x0[n], x1[n]);
    }
    float2* Z = fft_stockham<512, 64>(bufA, bufB, t, -1.f);

    // untangle packed spectra at wf = t (0..63)
    const float2 zk = Z[t];
    const float2 zm = Z[(512 - t) & 511];
    float4 o;
    o.x = 0.5f * (zk.x + zm.x);   // F_row0 re
    o.y = 0.5f * (zk.y - zm.y);   // F_row0 im
    o.z = 0.5f * (zk.y + zm.y);   // F_row1 re
    o.w = 0.5f * (zm.x - zk.x);   // F_row1 im
    *(float4*)(sm + ((size_t)(plane * 64 + t) * 256 + h0) * 2) = o;
}

// ---------------------------------------------------------------------------
// Kernel C: per (plane, wf<64) column: FFT_H(256) -> * filt -> iFFT_H, /256.
// ---------------------------------------------------------------------------
__global__ __launch_bounds__(64)
void colfft_kernel(float* __restrict__ sm, const float* __restrict__ filt) {
    __shared__ float2 bufA[256], bufB[256];
    const int t     = threadIdx.x;
    const int bid   = blockIdx.x;
    const int plane = bid >> 6;            // b*64 + c
    const int wf    = bid & 63;
    const int b     = plane >> 6;
    float2*       col = (float2*)sm + ((size_t)plane * 64 + wf) * 256;
    const float2* fl  = (const float2*)filt + ((size_t)b * 64 + wf) * 256;

    for (int i = 0; i < 4; ++i) bufA[t + 64 * i] = col[t + 64 * i];
    float2* Z = fft_stockham<256, 64>(bufA, bufB, t, -1.f);
    for (int i = 0; i < 4; ++i) {
        const int k = t + 64 * i;
        Z[k] = cmulf(Z[k], fl[k]);
    }
    float2* other = (Z == bufA) ? bufB : bufA;
    float2* Y = fft_stockham<256, 64>(Z, other, t, +1.f);
    const float sc = 1.f / 256.f;
    for (int i = 0; i < 4; ++i) {
        const int k = t + 64 * i;
        col[k] = make_float2(Y[k].x * sc, Y[k].y * sc);
    }
}

// ---------------------------------------------------------------------------
// Kernel D: recompute row FFT, splice modified wf<64 bins (Hermitian mirror,
// DC imag dropped), packed inverse 512-pt FFT, /512, write output rows.
// ---------------------------------------------------------------------------
__global__ __launch_bounds__(64)
void ifft_kernel(const float* __restrict__ x, const float* __restrict__ sm,
                 float* __restrict__ out) {
    __shared__ float2 bufA[512], bufB[512];
    const int t     = threadIdx.x;
    const int p     = blockIdx.x;
    const int plane = p >> 7;
    const int h0    = (p & 127) * 2;
    const float* x0 = x + ((size_t)plane * 256 + h0) * 512;
    const float* x1 = x0 + 512;
    for (int i = 0; i < 8; ++i) {
        const int n = t + 64 * i;
        bufA[n] = make_float2(x0[n], x1[n]);
    }
    float2* Z = fft_stockham<512, 64>(bufA, bufB, t, -1.f);
    float2* Q = (Z == bufA) ? bufB : bufA;
    for (int i = 0; i < 8; ++i) {
        const int n = t + 64 * i;
        Q[n] = Z[n];
    }
    __syncthreads();

    // splice in modified low bins: Q[k] = G1 + i*G2, Q[512-k] = conj(G1)+i*conj(G2)
    const float4 g = *(const float4*)(sm + ((size_t)(plane * 64 + t) * 256 + h0) * 2);
    float2 G1 = make_float2(g.x, g.y);
    float2 G2 = make_float2(g.z, g.w);
    if (t == 0) { G1.y = 0.f; G2.y = 0.f; }   // irfft ignores imag of DC bin
    Q[t] = make_float2(G1.x - G2.y, G1.y + G2.x);
    if (t != 0) Q[512 - t] = make_float2(G1.x + G2.y, G2.x - G1.y);

    float2* other = (Q == bufA) ? bufB : bufA;
    float2* Y = fft_stockham<512, 64>(Q, other, t, +1.f);
    const float sc = 1.f / 512.f;
    float* o0 = out + ((size_t)plane * 256 + h0) * 512;
    float* o1 = o0 + 512;
    for (int i = 0; i < 8; ++i) {
        const int n = t + 64 * i;
        o0[n] = Y[n].x * sc;
        o1[n] = Y[n].y * sc;
    }
}

// ---------------------------------------------------------------------------
extern "C" void kernel_launch(void* const* d_in, const int* in_sizes, int n_in,
                              void* d_out, int out_size, void* d_ws, size_t ws_size,
                              hipStream_t stream) {
    (void)in_sizes; (void)n_in; (void)out_size; (void)ws_size;
    const float* x  = (const float*)d_in[0];
    const float* dt = (const float*)d_in[1];
    const float* w  = (const float*)d_in[2];
    const float* bb = (const float*)d_in[3];
    float* out  = (float*)d_out;
    float* filt = (float*)d_ws;                                  // 1 MB
    float* sm   = filt + (size_t)BATCH * WFREQ * HDIM * 2;       // 64 MB

    filt_kernel  <<<BATCH * HDIM,             256, 0, stream>>>(x, dt, w, bb, filt);
    rowfft_kernel<<<BATCH * IN_CH * HDIM / 2,  64, 0, stream>>>(x, sm);
    colfft_kernel<<<BATCH * IN_CH * WFREQ,     64, 0, stream>>>(sm, filt);
    ifft_kernel  <<<BATCH * IN_CH * HDIM / 2,  64, 0, stream>>>(x, sm, out);
}
